// GaussianMILVariant_14465449853439
// MI455X (gfx1250) — matrix operand order repaired and checked by reading, not compiled, not run on hardware
//
#include <hip/hip_runtime.h>
#include <hip/hip_bf16.h>

typedef __attribute__((ext_vector_type(16))) __bf16 v16bf;
typedef __attribute__((ext_vector_type(8)))  float  v8f;

constexpr int kD    = 256;
constexpr int kS    = 512;
constexpr int kH    = 128;
constexpr int kGATE = 517;   // 2*D + 1 + 1 + 3
constexpr int kTM   = 16;    // tokens per block (WMMA M)
constexpr int kB    = 256;

__device__ __forceinline__ float sigmoidf_(float x) { return 1.f / (1.f + expf(-x)); }

// ---------------------------------------------------------------------------
// Kernel 0: repack W1 (f32 [128][517]) -> bf16 WMMA B-fragments for K=0..511.
// B-fragment (dense 16-bit 32x16): lane l: n = l&15, half = l>>4,
// element e (0..15): k = kt*32 + half*8 + (e&7) + (e>=8 ? 16:0).
// Storage: w1f[((kt*8 + nt)*32 + lane)*16 + e]
// ---------------------------------------------------------------------------
__global__ void __launch_bounds__(256)
repack_w1_kernel(const float* __restrict__ W1, __bf16* __restrict__ w1f) {
    int idx  = blockIdx.x * 256 + threadIdx.x;      // 0 .. 65535
    int e    = idx & 15;
    int lane = (idx >> 4) & 31;
    int tile = idx >> 9;                            // kt*8 + nt
    int nt   = tile & 7;
    int kt   = tile >> 3;
    int half = lane >> 4;
    int nloc = lane & 15;
    int k    = kt * 32 + half * 8 + (e & 7) + ((e >> 3) << 4);
    int ng   = nt * 16 + nloc;
    w1f[idx] = (__bf16)W1[ng * kGATE + k];
}

// ---------------------------------------------------------------------------
// Kernel 1: per-token reductions + gate MLP via WMMA.
// Block = 256 threads (8 waves) handles 16 tokens.
// ---------------------------------------------------------------------------
__global__ void __launch_bounds__(256)
token_kernel(const float* __restrict__ mu,
             const float* __restrict__ logvar,
             const float* __restrict__ minutes,
             const float* __restrict__ mu_c,
             const float* __restrict__ log_tau2,
             const float* __restrict__ log_prior,
             const float* __restrict__ W1,
             const float* __restrict__ b1,
             const float* __restrict__ W2,
             const float* __restrict__ b2,
             const __bf16* __restrict__ w1f,
             float* __restrict__ s_out,      // [B*S]
             float* __restrict__ score_ws,   // [B*S]
             float* __restrict__ araw_ws)    // [B*S]
{
    // A-fragments: [kt (16)][lane (32)][e (16)] bf16 = 16 KB
    __shared__ __align__(32) __bf16 afrag[16 * 32 * 16];
    __shared__ float stok[kTM], mint[kTM], sn1[kTM], sn2[kTM], sn3[kTM];
    __shared__ float hsum[kTM];

    const int tid     = threadIdx.x;
    const int tokBase = blockIdx.x * kTM;

    if (tid < kTM) hsum[tid] = 0.f;

    // ---------------- Phase A: load mu/logvar once, reduce + pack ----------
    {
        const int t   = tid >> 4;    // local token 0..15
        const int dg  = tid & 15;    // D-group: covers d = dg*16 .. dg*16+15
        const int tok = tokBase + t;
        const float* murow = mu     + (size_t)tok * kD;
        const float* lvrow = logvar + (size_t)tok * kD;

        float acc0 = 0.f, acc1 = 0.f, amu2 = 0.f, alv = 0.f, arat = 0.f;
        #pragma unroll
        for (int i = 0; i < 16; ++i) {
            int d = dg * 16 + i;
            float m   = murow[d];
            float lv  = lvrow[d];
            float var = expf(lv);
            float lt0 = log_tau2[d],      lt1 = log_tau2[kD + d];
            float d0  = m - mu_c[d],      d1  = m - mu_c[kD + d];
            acc0 += (var + d0 * d0) / expf(lt0) + lt0;
            acc1 += (var + d1 * d1) / expf(lt1) + lt1;
            amu2 += m * m;
            alv  += lv;
            arat += m * m / fmaxf(var, 1e-6f);

            // pack bf16 features into WMMA A-fragment layout.
            // A (16-bit 16x32): lane = half*16 + m ; k = kt*32 + half*8 + e + (e>=8 ? 8:0)
            // inverse: r = k&31 ; half = (r>>3)&1 ; e = (r&7) + ((r&16)>>1)
            {
                int k = d;                     // mu features: k = 0..255
                int kt = k >> 5, r = k & 31;
                int half = (r >> 3) & 1;
                int e = (r & 7) + ((r & 16) >> 1);
                afrag[((kt * 32) + half * 16 + t) * 16 + e] = (__bf16)m;
            }
            {
                int k = kD + d;                // logvar features: k = 256..511
                int kt = k >> 5, r = k & 31;
                int half = (r >> 3) & 1;
                int e = (r & 7) + ((r & 16) >> 1);
                afrag[((kt * 32) + half * 16 + t) * 16 + e] = (__bf16)lv;
            }
        }
        // reduce across the 16 lanes that share a token
        #pragma unroll
        for (int msk = 8; msk >= 1; msk >>= 1) {
            acc0 += __shfl_xor(acc0, msk, 16);
            acc1 += __shfl_xor(acc1, msk, 16);
            amu2 += __shfl_xor(amu2, msk, 16);
            alv  += __shfl_xor(alv,  msk, 16);
            arat += __shfl_xor(arat, msk, 16);
        }
        if (dg == 0) {
            float ell0 = -0.5f * acc0;
            float ell1 = -0.5f * acc1;
            float sv   = (ell1 + log_prior[1]) - (ell0 + log_prior[0]);
            stok[t] = sv;
            mint[t] = minutes[tok];
            sn1[t]  = sqrtf(amu2) * (1.0f / 16.0f);   // norm/sqrt(256)
            sn2[t]  = alv  * (1.0f / 256.0f);
            sn3[t]  = arat * (1.0f / 256.0f);
            s_out[tok]    = sv;
            score_ws[tok] = arat;
        }
    }
    __syncthreads();

    // ---------------- Phase B: GEMM  h[16 x 128] = feat[16 x 512] * W1^T ----
    const int wv   = tid >> 5;     // wave id = N tile (0..7)
    const int lane = tid & 31;

    v8f c = {0.f, 0.f, 0.f, 0.f, 0.f, 0.f, 0.f, 0.f};
    #pragma unroll 4
    for (int kt = 0; kt < 16; ++kt) {
        v16bf a  = *(const v16bf*)&afrag[(kt * 32 + lane) * 16];
        v16bf bm = *(const v16bf*)&w1f[(size_t)((kt * 8 + wv) * 32 + lane) * 16];
        c = __builtin_amdgcn_wmma_f32_16x16x32_bf16(
                false, a, false, bm, (short)0, c, false, false);
    }

    // tail features (f32 precision) + bias + relu + W2 dot
    const int half = lane >> 4;
    const int nloc = lane & 15;
    const int ng   = wv * 16 + nloc;
    const float tw_s  = W1[ng * kGATE + 512];
    const float tw_mn = W1[ng * kGATE + 513];
    const float tw_1  = W1[ng * kGATE + 514];
    const float tw_2  = W1[ng * kGATE + 515];
    const float tw_3  = W1[ng * kGATE + 516];
    const float bias  = b1[ng];
    const float w2v   = W2[ng];

    float part[8];
    #pragma unroll
    for (int r = 0; r < 8; ++r) {
        int m = r + half * 8;   // C/D layout: VGPR r, lanes 16-31 -> M = r + 8
        float h = c[r] + tw_s * stok[m] + tw_mn * mint[m]
                       + tw_1 * sn1[m] + tw_2 * sn2[m] + tw_3 * sn3[m] + bias;
        h = fmaxf(h, 0.f);
        part[r] = h * w2v;
    }
    // reduce over 16 n-columns (lanes within a half), accumulate across waves
    #pragma unroll
    for (int r = 0; r < 8; ++r) {
        float p = part[r];
        #pragma unroll
        for (int msk = 8; msk >= 1; msk >>= 1) p += __shfl_xor(p, msk, 16);
        if (nloc == 0) atomicAdd(&hsum[r + half * 8], p);
    }
    __syncthreads();

    if (tid < kTM) {
        araw_ws[tokBase + tid] = sigmoidf_(hsum[tid] + b2[0]);
    }
}

// ---------------------------------------------------------------------------
// Kernel 2: per-batch aggregation (softmax, mixing, time decay, noisy-OR).
// One block (512 threads) per batch row.
// ---------------------------------------------------------------------------
__device__ __forceinline__ float blockSum512(float v, float* buf) {
    int tid = threadIdx.x;
    buf[tid] = v; __syncthreads();
    #pragma unroll
    for (int off = 256; off >= 1; off >>= 1) {
        if (tid < off) buf[tid] += buf[tid + off];
        __syncthreads();
    }
    float r = buf[0]; __syncthreads();
    return r;
}
__device__ __forceinline__ float blockMax512(float v, float* buf) {
    int tid = threadIdx.x;
    buf[tid] = v; __syncthreads();
    #pragma unroll
    for (int off = 256; off >= 1; off >>= 1) {
        if (tid < off) buf[tid] = fmaxf(buf[tid], buf[tid + off]);
        __syncthreads();
    }
    float r = buf[0]; __syncthreads();
    return r;
}

__global__ void __launch_bounds__(512)
agg_kernel(const float* __restrict__ minutes,
           const float* __restrict__ s_vals,
           const float* __restrict__ score_ws,
           const float* __restrict__ araw_ws,
           const float* __restrict__ alpha_logits,
           const float* __restrict__ attn_lambda_logit,
           const float* __restrict__ decay_rate_log,
           float* __restrict__ s_total,
           float* __restrict__ a_out)
{
    __shared__ float buf[512];
    const int tid = threadIdx.x;
    const int b   = blockIdx.x;
    const int tok = b * kS + tid;

    float sc = score_ws[tok];
    float ar = araw_ws[tok];
    float sv = s_vals[tok];
    float mn = minutes[tok];

    // precision-attention softmax
    float maxsc  = blockMax512(sc, buf);
    float ex     = expf(sc - maxsc);
    float sumE   = blockSum512(ex, buf);
    float a_prec = ex / sumE;

    // normalized MLP attention
    float sumAr = blockSum512(ar, buf);
    float a_mlp = ar / fmaxf(sumAr, 1e-6f);

    float lam = sigmoidf_(attn_lambda_logit[0]);
    float a0  = lam * a_mlp + (1.f - lam) * a_prec;

    // time decay
    float maxMn = blockMax512(mn, buf);
    float delta = fmaxf(maxMn - mn, 0.f) * (1.f / 60.f);
    float rate  = log1pf(expf(decay_rate_log[0]));     // softplus
    float wv    = a0 * expf(-rate * delta);
    float sumW  = blockSum512(wv, buf);
    float a     = wv / fmaxf(sumW, 1e-6f);

    // mean aggregator
    float sMain = blockSum512(a * sv, buf);

    // noisy-OR
    float p = sigmoidf_(sv) * a;
    p = fminf(fmaxf(p, 1e-6f), 1.f - 1e-6f);
    float l1m = blockSum512(log1pf(-p), buf);

    a_out[tok] = a;
    if (tid == 0) {
        float p_or = 1.f - expf(l1m);
        p_or = fminf(fmaxf(p_or, 1e-6f), 1.f - 1e-6f);
        float s_or  = logf(p_or) - log1pf(-p_or);
        float alpha = sigmoidf_(alpha_logits[0]);
        s_total[b]  = alpha * sMain + (1.f - alpha) * s_or;
    }
}

// ---------------------------------------------------------------------------
extern "C" void kernel_launch(void* const* d_in, const int* in_sizes, int n_in,
                              void* d_out, int out_size, void* d_ws, size_t ws_size,
                              hipStream_t stream) {
    (void)in_sizes; (void)n_in; (void)out_size; (void)ws_size;

    const float* mu                = (const float*)d_in[0];
    const float* logvar            = (const float*)d_in[1];
    const float* minutes           = (const float*)d_in[2];
    const float* mu_c              = (const float*)d_in[3];
    const float* log_tau2          = (const float*)d_in[4];
    const float* log_prior         = (const float*)d_in[5];
    const float* W1                = (const float*)d_in[6];
    const float* b1                = (const float*)d_in[7];
    const float* W2                = (const float*)d_in[8];
    const float* b2                = (const float*)d_in[9];
    const float* alpha_logits      = (const float*)d_in[10];
    const float* attn_lambda_logit = (const float*)d_in[11];
    const float* decay_rate_log    = (const float*)d_in[12];

    const int BS = kB * kS;

    float* out     = (float*)d_out;
    float* s_total = out;                 // [B]
    float* s_out   = out + kB;            // [B*S]
    float* a_out   = out + kB + BS;       // [B*S]

    char*   ws       = (char*)d_ws;
    __bf16* w1f      = (__bf16*)ws;                              // 65536 bf16 = 128 KiB
    float*  score_ws = (float*)(ws + 65536 * sizeof(__bf16));    // [B*S]
    float*  araw_ws  = (float*)(ws + 65536 * sizeof(__bf16) + (size_t)BS * sizeof(float));

    repack_w1_kernel<<<dim3(65536 / 256), dim3(256), 0, stream>>>(W1, w1f);

    token_kernel<<<dim3(BS / kTM), dim3(256), 0, stream>>>(
        mu, logvar, minutes, mu_c, log_tau2, log_prior,
        W1, b1, W2, b2, w1f, s_out, score_ws, araw_ws);

    agg_kernel<<<dim3(kB), dim3(512), 0, stream>>>(
        minutes, s_out, score_ws, araw_ws,
        alpha_logits, attn_lambda_logit, decay_rate_log,
        s_total, a_out);
}
